// DeepRetrievalModule_7662221656279
// MI455X (gfx1250) — compile-verified, matrix operand order checked by reading, not run on hardware
//
#include <hip/hip_runtime.h>
#include <hip/hip_bf16.h>
#include <math.h>

// ---------------------------------------------------------------------------
// MI455X (gfx1250) implementation.
// - All GEMMs: split-bf16 (hi+lo, 3-term) -> ~fp32 accuracy at bf16 WMMA rate
// - TDM (tensor_load_to_lds) streams double-buffered LDS tiles, drained with
//   s_wait_tensorcnt; waves read fragments from LDS via ds_load_b128.
// - Block tile 64(M) x 128(N), 8 waves, each wave 16x64 (12 WMMAs / K-step).
// ---------------------------------------------------------------------------

typedef __attribute__((ext_vector_type(16))) __bf16   v16bf;
typedef __attribute__((ext_vector_type(8)))  float    v8f;
typedef __attribute__((ext_vector_type(4)))  unsigned u32x4;
typedef __attribute__((ext_vector_type(8)))  int      i32x8;
typedef __attribute__((ext_vector_type(4)))  int      i32x4;

constexpr int    BQ     = 2048;    // queries
constexpr int    NC     = 16384;   // candidates
constexpr int    INDIM  = 512;     // input dim
constexpr int    OUTD   = 512;     // projected dim D
constexpr int    KTOP   = 128;     // top-k
constexpr float  SIM_SCALE = 22.62741699796952f;   // sqrt(512)

#if __has_builtin(__builtin_amdgcn_tensor_load_to_lds) && \
    __has_builtin(__builtin_amdgcn_s_wait_tensorcnt)
#define USE_TDM 1
#else
#define USE_TDM 0
#endif

__device__ __forceinline__ __bf16 f2bf(float x) { return (__bf16)x; }

// ---------------------------------------------------------------------------
// Split f32 -> bf16 hi + bf16 residual lo
// ---------------------------------------------------------------------------
__global__ void split_bf16_kernel(const float* __restrict__ x,
                                  __bf16* __restrict__ h,
                                  __bf16* __restrict__ l, size_t n) {
  size_t i = (size_t)blockIdx.x * blockDim.x + threadIdx.x;
  size_t stride = (size_t)gridDim.x * blockDim.x;
  for (; i < n; i += stride) {
    float v  = x[i];
    __bf16 hi = f2bf(v);
    h[i] = hi;
    l[i] = f2bf(v - (float)hi);
  }
}

// ---------------------------------------------------------------------------
// TDM: 2-D tile load Global -> LDS. D# built per ISA 08_async_tensor §8:
//  group0: count=1 | lds_addr | global_addr(57b) | type=2
//  group1: data_size=1(2B) | tensor_dim0=K | tensor_dim1=rows_total |
//          tile_dim0=32 | tile_dim1=rows | tensor_dim0_stride=K
// This toolchain exposes the 6-arg builtin (extra int32x8 group + cpol).
// ---------------------------------------------------------------------------
__device__ __forceinline__ void tdm_issue_2d(const __bf16* gsrc,
                                             __bf16* ldst,
                                             int rows, int K, int tensor_rows) {
#if USE_TDM
  unsigned long long ga = (unsigned long long)(const void*)gsrc;
  unsigned lds_off = (unsigned)(unsigned long long)(void*)ldst;
  u32x4 g0;
  g0[0] = 1u;                                        // count=1, user mode
  g0[1] = lds_off;                                   // LDS byte address
  g0[2] = (unsigned)(ga & 0xffffffffull);            // global_addr[31:0]
  g0[3] = (unsigned)((ga >> 32) & 0x01ffffffull)     // global_addr[56:32]
        | (2u << 30);                                // type=2 (image)
  unsigned td0 = (unsigned)K;
  unsigned td1 = (unsigned)tensor_rows;
  unsigned long long s0 = (unsigned long long)K;     // dim0 stride (elems)
  i32x8 g1;
  g1[0] = (int)(1u << 16);                           // data_size=1 -> 2 bytes
  g1[1] = (int)((td0 & 0xffffu) << 16);              // tensor_dim0[15:0]
  g1[2] = (int)((td0 >> 16) | ((td1 & 0xffffu) << 16));
  g1[3] = (int)((td1 >> 16) | (32u << 16));          // tile_dim0 = 32
  g1[4] = (int)(unsigned)rows;                       // tile_dim1, tile_dim2=0
  g1[5] = (int)(unsigned)(s0 & 0xffffffffull);       // dim0_stride[31:0]
  g1[6] = (int)(unsigned)((s0 >> 32) & 0xffffull);   // dim0_stride[47:32]
  g1[7] = 0;
  i32x4 z4 = {0, 0, 0, 0};
  i32x8 z8 = {0, 0, 0, 0, 0, 0, 0, 0};
  __builtin_amdgcn_tensor_load_to_lds(g0, g1, z4, z4, z8, 0);
#else
  // Cooperative fallback: all 256 threads copy dwords (32 elems/row tile).
  const unsigned* src = (const unsigned*)(const void*)gsrc;
  unsigned* dst = (unsigned*)(void*)ldst;
  int ndw = rows * 16;                               // 32 bf16 = 16 dwords/row
  int ldw = K >> 1;
  for (int i = threadIdx.x; i < ndw; i += 256) {
    int r = i >> 4, c = i & 15;
    dst[i] = src[(size_t)r * ldw + c];
  }
  (void)tensor_rows;
#endif
}

// ---------------------------------------------------------------------------
// Load one 16x32 bf16 WMMA fragment from a row-major LDS tile (ld = 32).
// Lane l: row = row0 + (l&15); K-half = (l>>4)*8. VGPRs j=0..3 are K pairs
// {0..7}+half (16 contiguous bytes), j=4..7 are {16..23}+half -> 2x b128.
// ---------------------------------------------------------------------------
__device__ __forceinline__ v16bf lds_frag(const __bf16* tile, int row0, int lane) {
  const __bf16* p = tile + (size_t)(row0 + (lane & 15)) * 32 + ((lane >> 4) << 3);
  union { v16bf v; u32x4 q[2]; } f;
  f.q[0] = *reinterpret_cast<const u32x4*>(p);
  f.q[1] = *reinterpret_cast<const u32x4*>(p + 16);
  return f.v;
}

__device__ __forceinline__ v8f wmma_bf16(v16bf a, v16bf b, v8f c) {
  return __builtin_amdgcn_wmma_f32_16x16x32_bf16(
      false, a, false, b, (short)0, c, false, false);
}

// ---------------------------------------------------------------------------
// C[M,N] = A[M,K]*B[N,K]^T (split-bf16, 3-term). TDM double-buffered tiles.
// Epilogue modes: 0: outF=acc+bias[n]; 1: outH/L=split(acc+bias[n]);
//                 2: outF = -(sq[m]+sc[n]-2*acc)*sqrt(D)
// Requires M%64==0, N%128==0, K%32==0.
// ---------------------------------------------------------------------------
__global__ void __launch_bounds__(256)
gemm_tdm_wmma_kernel(const __bf16* __restrict__ Ah, const __bf16* __restrict__ Al,
                     int M,
                     const __bf16* __restrict__ Bh, const __bf16* __restrict__ Bl,
                     int N, int K, int mode,
                     const float* __restrict__ bias,
                     float* __restrict__ outF,
                     __bf16* __restrict__ outH, __bf16* __restrict__ outL,
                     const float* __restrict__ sq, const float* __restrict__ sc) {
  __shared__ __align__(16) __bf16 sAh[2][64 * 32];
  __shared__ __align__(16) __bf16 sAl[2][64 * 32];
  __shared__ __align__(16) __bf16 sBh[2][128 * 32];
  __shared__ __align__(16) __bf16 sBl[2][128 * 32];

  const int lane = threadIdx.x & 31;
  const int wid  = threadIdx.x >> 5;      // 0..7
  const int wm   = wid & 3;               // M sub-tile (16 rows)
  const int wn   = wid >> 2;              // N sub-tile (64 cols)
  const int bm   = blockIdx.x % (M >> 6);
  const int bn   = blockIdx.x / (M >> 6);

  const __bf16* gAh = Ah + (size_t)bm * 64 * K;
  const __bf16* gAl = Al + (size_t)bm * 64 * K;
  const __bf16* gBh = Bh + (size_t)bn * 128 * K;
  const __bf16* gBl = Bl + (size_t)bn * 128 * K;

  auto issue_tiles = [&](int buf, int k0) {
#if USE_TDM
    if (wid == 0) {                        // one wave drives the TDM
      tdm_issue_2d(gAh + k0, sAh[buf], 64, K, M);
      tdm_issue_2d(gAl + k0, sAl[buf], 64, K, M);
      tdm_issue_2d(gBh + k0, sBh[buf], 128, K, N);
      tdm_issue_2d(gBl + k0, sBl[buf], 128, K, N);
    }
#else
    tdm_issue_2d(gAh + k0, sAh[buf], 64, K, M);
    tdm_issue_2d(gAl + k0, sAl[buf], 64, K, M);
    tdm_issue_2d(gBh + k0, sBh[buf], 128, K, N);
    tdm_issue_2d(gBl + k0, sBl[buf], 128, K, N);
#endif
  };

  v8f acc[4] = {};
  const int nsteps = K >> 5;

  issue_tiles(0, 0);
  for (int step = 0; step < nsteps; ++step) {
    const int cur = step & 1;
    const bool more = (step + 1) < nsteps;
    if (more) issue_tiles(cur ^ 1, (step + 1) << 5);
#if USE_TDM
    if (wid == 0) {
      if (more) __builtin_amdgcn_s_wait_tensorcnt(4);  // current 4 tiles done
      else      __builtin_amdgcn_s_wait_tensorcnt(0);
    }
#endif
    __syncthreads();                       // tiles[cur] visible to all waves

    v16bf ah = lds_frag(sAh[cur], wm * 16, lane);
    v16bf al = lds_frag(sAl[cur], wm * 16, lane);
#pragma unroll
    for (int s = 0; s < 4; ++s) {
      v16bf bh = lds_frag(sBh[cur], wn * 64 + s * 16, lane);
      v16bf bl = lds_frag(sBl[cur], wn * 64 + s * 16, lane);
      acc[s] = wmma_bf16(ah, bh, acc[s]);  // hi*hi
      acc[s] = wmma_bf16(ah, bl, acc[s]);  // hi*lo
      acc[s] = wmma_bf16(al, bh, acc[s]);  // lo*hi
    }
    __syncthreads();                       // safe to overwrite tiles[cur]
  }

  // C/D layout: VGPR r -> M = r (+8 for lanes 16..31), N = lane&15
  const int mrow = (lane >> 4) << 3;
  const int n0   = lane & 15;
#pragma unroll
  for (int s = 0; s < 4; ++s) {
#pragma unroll
    for (int r = 0; r < 8; ++r) {
      int m = bm * 64 + wm * 16 + mrow + r;
      int n = bn * 128 + wn * 64 + s * 16 + n0;
      float a = acc[s][r];
      size_t idx = (size_t)m * N + n;
      if (mode == 0) {
        outF[idx] = a + bias[n];
      } else if (mode == 1) {
        float p = a + bias[n];
        __bf16 hi = f2bf(p);
        outH[idx] = hi;
        outL[idx] = f2bf(p - (float)hi);
      } else {
        outF[idx] = -(sq[m] + sc[n] - 2.0f * a) * SIM_SCALE;
      }
    }
  }
}

// ---------------------------------------------------------------------------
// Row squared-norms of a split-bf16 matrix: out[row] = sum((hi+lo)^2)
// ---------------------------------------------------------------------------
__global__ void rownorm_kernel(const __bf16* __restrict__ h,
                               const __bf16* __restrict__ l,
                               float* __restrict__ out, int K) {
  const int row = blockIdx.x;
  const __bf16* ph = h + (size_t)row * K;
  const __bf16* pl = l + (size_t)row * K;
  float s = 0.0f;
  for (int i = threadIdx.x; i < K; i += blockDim.x) {
    float v = (float)ph[i] + (float)pl[i];
    s += v * v;
  }
  __shared__ float red[128];
  red[threadIdx.x] = s;
  __syncthreads();
  for (int o = blockDim.x >> 1; o > 0; o >>= 1) {
    if (threadIdx.x < (unsigned)o) red[threadIdx.x] += red[threadIdx.x + o];
    __syncthreads();
  }
  if (threadIdx.x == 0) out[row] = red[0];
}

// ---------------------------------------------------------------------------
// Softmax row statistics: max and sum(exp(x - max)) over a sim row.
// ---------------------------------------------------------------------------
__global__ void row_stats_kernel(const float* __restrict__ sim,
                                 float* __restrict__ rowmax,
                                 float* __restrict__ rowsum) {
  const int b = blockIdx.x;
  const float* row = sim + (size_t)b * NC;
  __shared__ float red[256];
  float m = -INFINITY;
  for (int i = threadIdx.x; i < NC; i += 256) m = fmaxf(m, row[i]);
  red[threadIdx.x] = m;
  __syncthreads();
  for (int o = 128; o > 0; o >>= 1) {
    if (threadIdx.x < (unsigned)o)
      red[threadIdx.x] = fmaxf(red[threadIdx.x], red[threadIdx.x + o]);
    __syncthreads();
  }
  m = red[0];
  __syncthreads();
  float s = 0.0f;
  for (int i = threadIdx.x; i < NC; i += 256) s += __expf(row[i] - m);
  red[threadIdx.x] = s;
  __syncthreads();
  for (int o = 128; o > 0; o >>= 1) {
    if (threadIdx.x < (unsigned)o) red[threadIdx.x] += red[threadIdx.x + o];
    __syncthreads();
  }
  if (threadIdx.x == 0) { rowmax[b] = m; rowsum[b] = red[0]; }
}

// ---------------------------------------------------------------------------
// Exact top-128 per row via 2-level radix select on monotonic uint keys,
// then O(k^2) rank sort (descending). Writes sorted indices + softmax values.
// ---------------------------------------------------------------------------
__device__ __forceinline__ unsigned mono_key(float f) {
  unsigned s = __float_as_uint(f);
  return (s & 0x80000000u) ? ~s : (s | 0x80000000u);
}

__global__ void topk_kernel(const float* __restrict__ sim,
                            const float* __restrict__ rowmax,
                            const float* __restrict__ rowsum,
                            int* __restrict__ topk_idx,
                            float* __restrict__ topk_val) {
  const int b = blockIdx.x;
  const float* row = sim + (size_t)b * NC;
  const int tid = threadIdx.x;

  __shared__ unsigned hist[256];
  __shared__ unsigned svals[KTOP];
  __shared__ int      sidx[KTOP];
  __shared__ unsigned selCnt, tieCnt;
  __shared__ int sT, sT2, sNeed, sNeed2;

  // pass 1: histogram of top byte
  hist[tid] = 0;
  __syncthreads();
  for (int i = tid; i < NC; i += 256)
    atomicAdd(&hist[mono_key(row[i]) >> 24], 1u);
  __syncthreads();
  if (tid == 0) {
    unsigned acc = 0; int t = 0;
    for (int bb = 255; bb >= 0; --bb) {
      if (acc + hist[bb] >= (unsigned)KTOP) { t = bb; break; }
      acc += hist[bb];
    }
    sT = t; sNeed = KTOP - (int)acc;
  }
  __syncthreads();

  // pass 2: histogram of second byte within threshold bin
  hist[tid] = 0;
  __syncthreads();
  for (int i = tid; i < NC; i += 256) {
    unsigned u = mono_key(row[i]);
    if ((int)(u >> 24) == sT) atomicAdd(&hist[(u >> 16) & 255], 1u);
  }
  __syncthreads();
  if (tid == 0) {
    unsigned acc = 0; int t2 = 0;
    for (int bb = 255; bb >= 0; --bb) {
      if (acc + hist[bb] >= (unsigned)sNeed) { t2 = bb; break; }
      acc += hist[bb];
    }
    sT2 = t2; sNeed2 = sNeed - (int)acc;
    selCnt = 0; tieCnt = 0;
  }
  __syncthreads();

  // pass 3: collect exactly 128 winners
  for (int i = tid; i < NC; i += 256) {
    unsigned u = mono_key(row[i]);
    int hi = (int)(u >> 24);
    bool sel = false;
    if (hi > sT) sel = true;
    else if (hi == sT) {
      int mid = (int)((u >> 16) & 255);
      if (mid > sT2) sel = true;
      else if (mid == sT2) {
        unsigned tp = atomicAdd(&tieCnt, 1u);
        if (tp < (unsigned)sNeed2) sel = true;
      }
    }
    if (sel) {
      unsigned p = atomicAdd(&selCnt, 1u);
      svals[p] = u;
      sidx[p]  = i;
    }
  }
  __syncthreads();

  // pass 4: rank sort descending (ties -> earlier slot first), write outputs
  if (tid < KTOP) {
    unsigned my = svals[tid];
    int rank = 0;
    for (int j = 0; j < KTOP; ++j) {
      unsigned o = svals[j];
      if (o > my || (o == my && j < tid)) ++rank;
    }
    int idx = sidx[tid];
    float attn = __expf(row[idx] - rowmax[b]) / rowsum[b];
    topk_idx[(size_t)b * KTOP + rank] = idx;
    topk_val[(size_t)b * KTOP + rank] = attn;
  }
}

// ---------------------------------------------------------------------------
// Gather: out[pair, 0:512] = v[topk_idx[pair], 0:512]; 2KB rows via float4.
// ---------------------------------------------------------------------------
__global__ void gather_kernel(const float* __restrict__ v,
                              const int* __restrict__ topk_idx,
                              float* __restrict__ out) {
  const int pair = blockIdx.x;             // b*KTOP + j
  const int row  = topk_idx[pair];
  const float4* src = reinterpret_cast<const float4*>(v + (size_t)row * OUTD);
  float4* dst = reinterpret_cast<float4*>(out + (size_t)pair * OUTD);
  __builtin_prefetch(src + threadIdx.x, 0, 0);   // global_prefetch_b8
  dst[threadIdx.x] = src[threadIdx.x];
}

// ---------------------------------------------------------------------------
extern "C" void kernel_launch(void* const* d_in, const int* in_sizes, int n_in,
                              void* d_out, int out_size, void* d_ws, size_t ws_size,
                              hipStream_t stream) {
  (void)in_sizes; (void)n_in; (void)out_size; (void)ws_size;
  const float* query = (const float*)d_in[0];
  const float* cand  = (const float*)d_in[1];
  const float* Kw    = (const float*)d_in[2];
  const float* Kb    = (const float*)d_in[3];
  const float* Vw    = (const float*)d_in[4];
  const float* Vb    = (const float*)d_in[5];

  float* out      = (float*)d_out;
  float* out_vals = out + (size_t)BQ * KTOP * OUTD;

  char* w = (char*)d_ws;
  size_t off = 0;
  auto alloc = [&](size_t bytes) {
    void* p = w + off;
    off = (off + bytes + 255) & ~(size_t)255;
    return p;
  };

  __bf16* qh  = (__bf16*)alloc((size_t)BQ * INDIM * 2);
  __bf16* ql  = (__bf16*)alloc((size_t)BQ * INDIM * 2);
  __bf16* ch  = (__bf16*)alloc((size_t)NC * INDIM * 2);
  __bf16* cl  = (__bf16*)alloc((size_t)NC * INDIM * 2);
  __bf16* kwh = (__bf16*)alloc((size_t)OUTD * INDIM * 2);
  __bf16* kwl = (__bf16*)alloc((size_t)OUTD * INDIM * 2);
  __bf16* vwh = (__bf16*)alloc((size_t)OUTD * INDIM * 2);
  __bf16* vwl = (__bf16*)alloc((size_t)OUTD * INDIM * 2);
  __bf16* kqh = (__bf16*)alloc((size_t)BQ * OUTD * 2);
  __bf16* kql = (__bf16*)alloc((size_t)BQ * OUTD * 2);
  __bf16* kch = (__bf16*)alloc((size_t)NC * OUTD * 2);
  __bf16* kcl = (__bf16*)alloc((size_t)NC * OUTD * 2);
  float*  vbuf   = (float*)alloc((size_t)NC * OUTD * 4);
  float*  sim    = (float*)alloc((size_t)BQ * NC * 4);
  float*  sq     = (float*)alloc((size_t)BQ * 4);
  float*  sc     = (float*)alloc((size_t)NC * 4);
  float*  rowmax = (float*)alloc((size_t)BQ * 4);
  float*  rowsum = (float*)alloc((size_t)BQ * 4);
  int*    tidx   = (int*)alloc((size_t)BQ * KTOP * 4);

  // 1) split inputs into bf16 hi/lo
  split_bf16_kernel<<<1024, 256, 0, stream>>>(query, qh, ql, (size_t)BQ * INDIM);
  split_bf16_kernel<<<4096, 256, 0, stream>>>(cand,  ch, cl, (size_t)NC * INDIM);
  split_bf16_kernel<<<512,  256, 0, stream>>>(Kw, kwh, kwl, (size_t)OUTD * INDIM);
  split_bf16_kernel<<<512,  256, 0, stream>>>(Vw, vwh, vwl, (size_t)OUTD * INDIM);

  // 2) projections (A * W^T + b) via TDM + WMMA
  gemm_tdm_wmma_kernel<<<(BQ / 64) * (OUTD / 128), 256, 0, stream>>>(
      qh, ql, BQ, kwh, kwl, OUTD, INDIM, /*mode=*/1, Kb,
      nullptr, kqh, kql, nullptr, nullptr);
  gemm_tdm_wmma_kernel<<<(NC / 64) * (OUTD / 128), 256, 0, stream>>>(
      ch, cl, NC, kwh, kwl, OUTD, INDIM, /*mode=*/1, Kb,
      nullptr, kch, kcl, nullptr, nullptr);
  gemm_tdm_wmma_kernel<<<(NC / 64) * (OUTD / 128), 256, 0, stream>>>(
      ch, cl, NC, vwh, vwl, OUTD, INDIM, /*mode=*/0, Vb,
      vbuf, nullptr, nullptr, nullptr, nullptr);

  // 3) squared norms of projected rows
  rownorm_kernel<<<BQ, 128, 0, stream>>>(kqh, kql, sq, OUTD);
  rownorm_kernel<<<NC, 128, 0, stream>>>(kch, kcl, sc, OUTD);

  // 4) sim = -(||kq||^2 + ||kc||^2 - 2 kq.kc) * sqrt(D)
  gemm_tdm_wmma_kernel<<<(BQ / 64) * (NC / 128), 256, 0, stream>>>(
      kqh, kql, BQ, kch, kcl, NC, OUTD, /*mode=*/2, nullptr,
      sim, nullptr, nullptr, sq, sc);

  // 5) softmax stats, top-k, gather
  row_stats_kernel<<<BQ, 256, 0, stream>>>(sim, rowmax, rowsum);
  topk_kernel<<<BQ, 256, 0, stream>>>(sim, rowmax, rowsum, tidx, out_vals);
  gather_kernel<<<BQ * KTOP, 128, 0, stream>>>(vbuf, tidx, out);
}
